// PointNetRes_75170517614893
// MI455X (gfx1250) — matrix operand order repaired and checked by reading, not compiled, not run on hardware
//
#include <hip/hip_runtime.h>
#include <hip/hip_bf16.h>

// ---------------------------------------------------------------------------
// PointNetRes forward for MI455X (gfx1250): all 1x1 convs + conv8 as f16 WMMA
// GEMMs (f32 accumulate), BN-train stats as reduction kernels, softpool as
// LDS top-32 selection. Activations live in (C x P) layout, P = B*N = 65536.
// GEMM: double-buffered LDS, 128-bit LDS accesses, 1 barrier per K-step.
// ---------------------------------------------------------------------------

typedef __attribute__((ext_vector_type(16))) _Float16 v16h;
typedef __attribute__((ext_vector_type(8)))  float    v8f;

#define NPTS   4096
#define BATCH  16
#define PCOLS  65536           // B * N
#define NEG_BIG (-3.402823466e38f)

union HFrag { v16h v; uint4 q[2]; _Float16 h[16]; };
union CFrag { v8f  v; float f[8]; };

// ---------------------------------------------------------------------------
// WMMA GEMM:  C(M x P) = A(M x K) * B(K x P) [+ bias per row]
// Block: 256 threads = 8 waves; tile 64(M) x 128(N); BK = 32; LDS ping-pong.
// Wave w: subtile row rm = w&3; 4 col subtiles cnB = (w>>2)*4 .. +3.
//   A tile LDS row-major 64x32: A-frag = two ds_load_b128
//     (lane g,r holds K = [g*16, g*16+15] of row rm*16+r).
//   B tile LDS fragment-major [cn(8)][lane(32)][kh(16)]: B-frag = two
//     ds_load_b128 (lane g*16+n holds K = g*16..g*16+15 of col cn*16+n).
// SPLITK variant writes partial tiles to C + z*M*P (no bias), deterministic.
// ---------------------------------------------------------------------------
template<bool SPLITK>
__global__ __launch_bounds__(256)
void wmma_gemm_kernel(const _Float16* __restrict__ A,
                      const _Float16* __restrict__ B,
                      float* __restrict__ C,
                      const float* __restrict__ bias,
                      int M, int K, int P, int kChunk)
{
    __shared__ __align__(16) _Float16 As[2][64 * 32];       // row-major  (2x4 KB)
    __shared__ __align__(16) _Float16 BsF[2][8 * 32 * 16];  // frag-major (2x8 KB)

    const int t    = threadIdx.x;
    const int wave = t >> 5;
    const int lid  = t & 31;
    const int g    = lid >> 4;      // lane group (0/1)
    const int r    = lid & 15;      // row-in-group / col-in-group
    const int rm   = wave & 3;
    const int cnB  = (wave >> 2) << 2;   // 0 or 4

    const int tileM = blockIdx.y * 64;
    const int tileN = blockIdx.x * 128;

    int k0 = 0, kEnd = K;
    if (SPLITK) {
        k0   = blockIdx.z * kChunk;
        kEnd = (k0 + kChunk < K) ? (k0 + kChunk) : K;
    }

    // ---- per-thread staging addresses (hoisted out of the K loop) ----
    // A: thread t copies row t>>2, halves (t&3)*8 .. +7
    const size_t aGlb = (size_t)(tileM + (t >> 2)) * K + (t & 3) * 8;
    // B: thread t handles two uint4 (s = 0,1): idx = s*256+t -> krow, seg
    int    bLds[2];
    size_t bGlb[2];
    #pragma unroll
    for (int s = 0; s < 2; ++s) {
        const int idx  = s * 256 + t;
        const int krow = idx >> 4;          // 0..31
        const int seg  = idx & 15;          // 8 cols each
        // frag-major scatter base: ((cn<<5) + laneBase) << 4) + kh,  elem e at +e*16
        bLds[s] = ((((seg >> 1) << 5) + ((krow >> 4) << 4) + ((seg & 1) << 3)) << 4)
                + (krow & 15);
        bGlb[s] = (size_t)krow * P + tileN + seg * 8;
    }

    v8f acc[4] = {};

    // ---- prologue: load + stage first tile into buffer 0 ----
    {
        const uint4 a0  = *(const uint4*)(A + aGlb + k0);
        const uint4 bb0 = *(const uint4*)(B + (size_t)k0 * P + bGlb[0]);
        const uint4 bb1 = *(const uint4*)(B + (size_t)k0 * P + bGlb[1]);
        ((uint4*)As[0])[t] = a0;
        const _Float16* h0 = (const _Float16*)&bb0;
        const _Float16* h1 = (const _Float16*)&bb1;
        #pragma unroll
        for (int e = 0; e < 8; ++e) {
            BsF[0][bLds[0] + (e << 4)] = h0[e];
            BsF[0][bLds[1] + (e << 4)] = h1[e];
        }
    }
    __syncthreads();

    int buf = 0;
    for (int kk = k0; kk < kEnd; kk += 32) {
        const bool more = (kk + 32) < kEnd;

        // ---- issue next tile's global loads early (overlap with WMMA) ----
        uint4 na, nb0, nb1;
        if (more) {
            na  = *(const uint4*)(A + aGlb + kk + 32);
            nb0 = *(const uint4*)(B + (size_t)(kk + 32) * P + bGlb[0]);
            nb1 = *(const uint4*)(B + (size_t)(kk + 32) * P + bGlb[1]);
        }
        if (kk + 64 < kEnd) {
            __builtin_prefetch(B + (size_t)(kk + 64) * P + bGlb[0], 0, 1);
            __builtin_prefetch(B + (size_t)(kk + 64) * P + bGlb[1], 0, 1);
        }

        // ---- A fragment: two contiguous 16B runs of its row ----
        HFrag af;
        {
            const _Float16* ap = &As[buf][(rm * 16 + r) * 32 + g * 16];
            af.q[0] = *(const uint4*)ap;          // K = g*16 .. g*16+7
            af.q[1] = *(const uint4*)(ap + 8);    // K = g*16+8 .. g*16+15
        }

        // ---- 4 B fragments + 4 WMMAs ----
        #pragma unroll
        for (int c = 0; c < 4; ++c) {
            HFrag bf;
            const _Float16* bp = &BsF[buf][(((cnB + c) << 5) + lid) << 4];
            bf.q[0] = *(const uint4*)bp;
            bf.q[1] = *(const uint4*)(bp + 8);
            acc[c] = __builtin_amdgcn_wmma_f32_16x16x32_f16(false, af.v, false, bf.v,
                                                            (short)0, acc[c], false, false);
        }

        // ---- stage next tile into the other buffer ----
        if (more) {
            ((uint4*)As[buf ^ 1])[t] = na;
            const _Float16* h0 = (const _Float16*)&nb0;
            const _Float16* h1 = (const _Float16*)&nb1;
            #pragma unroll
            for (int e = 0; e < 8; ++e) {
                BsF[buf ^ 1][bLds[0] + (e << 4)] = h0[e];
                BsF[buf ^ 1][bLds[1] + (e << 4)] = h1[e];
            }
        }
        __syncthreads();
        buf ^= 1;
    }

    // ---- C writeout: VGPR i -> row = i + 8*g, col = r ----
    float* Cz = SPLITK ? (C + (size_t)blockIdx.z * (size_t)M * (size_t)P) : C;
    #pragma unroll
    for (int c = 0; c < 4; ++c) {
        CFrag cf; cf.v = acc[c];
        const int colc = tileN + (cnB + c) * 16 + r;
        #pragma unroll
        for (int i = 0; i < 8; ++i) {
            const int row = tileM + rm * 16 + g * 8 + i;
            const size_t o = (size_t)row * P + colc;
            if (SPLITK) Cz[o] = cf.f[i];
            else        Cz[o] = cf.f[i] + (bias ? bias[row] : 0.f);
        }
    }
}

// ---------------------------------------------------------------------------
// Input pack: x (B,4,N) f32 -> X0 (32 x P) f16, rows 4..31 zero (K pad)
// ---------------------------------------------------------------------------
__global__ void pack_input_kernel(const float* __restrict__ x, _Float16* __restrict__ X0)
{
    const size_t i = (size_t)blockIdx.x * 256 + threadIdx.x;   // 32*P
    const int c = (int)(i >> 16), j = (int)(i & 65535);
    const int b = j >> 12, n = j & 4095;
    X0[i] = (c < 4) ? (_Float16)x[((size_t)b * 4 + c) * NPTS + n] : (_Float16)0.f;
}

// Weight convert + K-pad: W (O x C) f32 -> Wh (O x Kp) f16
__global__ void cvt_pad_kernel(const float* __restrict__ W, _Float16* __restrict__ Wh,
                               int O, int C, int Kp)
{
    const size_t i = (size_t)blockIdx.x * 256 + threadIdx.x;
    if (i >= (size_t)O * Kp) return;
    const int o = (int)(i / Kp), k = (int)(i % Kp);
    Wh[i] = (k < C) ? (_Float16)W[(size_t)o * C + k] : (_Float16)0.f;
}

// ---------------------------------------------------------------------------
// BN (training mode) per-channel stats over P columns -> scale/shift
// ---------------------------------------------------------------------------
__global__ __launch_bounds__(256)
void bn_stats_kernel(const float* __restrict__ H, const float* __restrict__ gam,
                     const float* __restrict__ bet, float* __restrict__ scale,
                     float* __restrict__ shift)
{
    const int ch = blockIdx.x, t = threadIdx.x;
    const float* row = H + (size_t)ch * PCOLS;
    float s = 0.f, s2 = 0.f;
    for (int i = t; i < PCOLS; i += 256) { const float v = row[i]; s += v; s2 += v * v; }
    __shared__ float rs[256], rq[256];
    rs[t] = s; rq[t] = s2; __syncthreads();
    for (int off = 128; off > 0; off >>= 1) {
        if (t < off) { rs[t] += rs[t + off]; rq[t] += rq[t + off]; }
        __syncthreads();
    }
    if (t == 0) {
        const float m   = rs[0] * (1.f / PCOLS);
        const float var = rq[0] * (1.f / PCOLS) - m * m;
        const float sc  = gam[ch] * rsqrtf(var + 1e-5f);
        scale[ch] = sc;
        shift[ch] = bet[ch] - m * sc;
    }
}

template<bool RELU, bool F16OUT>
__global__ void bn_apply_kernel(const float* __restrict__ H, const float* __restrict__ scale,
                                const float* __restrict__ shift, void* __restrict__ out)
{
    const size_t i = (size_t)blockIdx.x * 256 + threadIdx.x;
    const int ch = (int)(i >> 16);
    float v = H[i] * scale[ch] + shift[ch];
    if (RELU) v = fmaxf(v, 0.f);
    if (F16OUT) ((_Float16*)out)[i] = (_Float16)v;
    else        ((float*)out)[i]    = v;
}

// ---------------------------------------------------------------------------
// softpool part 1: top-32 indices (descending, lowest-index tie-break)
// grid = (256 sort channels, 16 batches); row of 4096 staged in LDS.
// ---------------------------------------------------------------------------
__global__ __launch_bounds__(256)
void topk32_kernel(const float* __restrict__ G3, int* __restrict__ idxOut)
{
    __shared__ float vals[NPTS];
    __shared__ float rv[256];
    __shared__ int   ri[256];
    const int c = blockIdx.x, b = blockIdx.y, t = threadIdx.x;
    const float* row = G3 + (size_t)c * PCOLS + (size_t)b * NPTS;
    for (int i = t; i < NPTS; i += 256) vals[i] = row[i];
    __syncthreads();

    for (int s = 0; s < 32; ++s) {
        float best = NEG_BIG; int bi = 0;
        for (int i = t; i < NPTS; i += 256) {
            const float v = vals[i];
            if (v > best) { best = v; bi = i; }
        }
        rv[t] = best; ri[t] = bi; __syncthreads();
        for (int off = 128; off > 0; off >>= 1) {
            if (t < off) {
                if (rv[t + off] > rv[t] ||
                    (rv[t + off] == rv[t] && ri[t + off] < ri[t])) {
                    rv[t] = rv[t + off]; ri[t] = ri[t + off];
                }
            }
            __syncthreads();
        }
        if (t == 0) {
            idxOut[((size_t)b * 256 + c) * 32 + s] = ri[0];
            vals[ri[0]] = NEG_BIG;
        }
        __syncthreads();
    }
}

// softpool part 2: SP (K=65536 x 512) f16, k = c*256 + w, col j = b*32 + h
__global__ void gather_sp_kernel(const float* __restrict__ G3, const int* __restrict__ idx,
                                 _Float16* __restrict__ SP)
{
    const size_t i = (size_t)blockIdx.x * 256 + threadIdx.x;   // 65536*512
    const int k = (int)(i >> 9), j = (int)(i & 511);
    const int c = k >> 8, w = k & 255;
    const int b = j >> 5, h = j & 31;
    const int n = idx[((size_t)b * 256 + w) * 32 + h];
    SP[i] = (_Float16)G3[(size_t)c * PCOLS + ((size_t)b << 12) + n];
}

// reduce split-K partials for conv8 + bias: Y8 (256 x 512)
__global__ void reduce_y8_kernel(const float* __restrict__ part, const float* __restrict__ b8,
                                 float* __restrict__ Y8, int nz)
{
    const int i = blockIdx.x * 256 + threadIdx.x;   // 131072
    float s = b8[i >> 9];
    for (int z = 0; z < nz; ++z) s += part[(size_t)z * 131072 + i];
    Y8[i] = s;
}

// conv9: glob[b][o] = b9[o] + sum_{c,h} Y8[c][b*32+h] * w9[o,c,h]
__global__ __launch_bounds__(256)
void conv9_kernel(const float* __restrict__ Y8, const float* __restrict__ w9,
                  const float* __restrict__ b9, float* __restrict__ glob)
{
    const int o = blockIdx.x, b = blockIdx.y, t = threadIdx.x;
    float s = 0.f;
    for (int k = t; k < 8192; k += 256) {
        const int c = k >> 5, h = k & 31;
        s += Y8[(size_t)c * 512 + b * 32 + h] * w9[(size_t)o * 8192 + k];
    }
    __shared__ float red[256];
    red[t] = s; __syncthreads();
    for (int off = 128; off > 0; off >>= 1) {
        if (t < off) red[t] += red[t + off];
        __syncthreads();
    }
    if (t == 0) glob[b * 256 + o] = red[0] + b9[o];
}

// X4 (320 x P) f16: rows 0..255 broadcast glob, rows 256..319 pointfeat
__global__ void build_x4_kernel(const float* __restrict__ glob, const _Float16* __restrict__ X1,
                                _Float16* __restrict__ X4)
{
    const size_t i = (size_t)blockIdx.x * 256 + threadIdx.x;   // 320*P
    const int ch = (int)(i >> 16), j = (int)(i & 65535);
    if (ch < 256) X4[i] = (_Float16)glob[(j >> 12) * 256 + ch];
    else          X4[i] = X1[(size_t)(ch - 256) * PCOLS + j];
}

// conv7 + tanh: out (B,3,N) f32 from X7 (128 x P) f16
__global__ __launch_bounds__(256)
void conv7_kernel(const _Float16* __restrict__ X7, const float* __restrict__ w7,
                  const float* __restrict__ b7, float* __restrict__ out)
{
    __shared__ float w[3 * 128];
    const int t = threadIdx.x;
    for (int i = t; i < 384; i += 256) w[i] = w7[i];
    __syncthreads();
    const size_t j = (size_t)blockIdx.x * 256 + t;   // P columns
    float a0 = b7[0], a1 = b7[1], a2 = b7[2];
    for (int c = 0; c < 128; ++c) {
        const float v = (float)X7[(size_t)c * PCOLS + j];
        a0 += v * w[c]; a1 += v * w[128 + c]; a2 += v * w[256 + c];
    }
    const int b = (int)(j >> 12), n = (int)(j & 4095);
    out[((size_t)b * 3 + 0) * NPTS + n] = tanhf(a0);
    out[((size_t)b * 3 + 1) * NPTS + n] = tanhf(a1);
    out[((size_t)b * 3 + 2) * NPTS + n] = tanhf(a2);
}

// ---------------------------------------------------------------------------
extern "C" void kernel_launch(void* const* d_in, const int* in_sizes, int n_in,
                              void* d_out, int out_size, void* d_ws, size_t ws_size,
                              hipStream_t stream)
{
    (void)in_sizes; (void)n_in; (void)out_size; (void)ws_size;
    const float* x   = (const float*)d_in[0];
    const float* w1  = (const float*)d_in[1];  const float* b1  = (const float*)d_in[2];
    const float* g1  = (const float*)d_in[3];  const float* be1 = (const float*)d_in[4];
    const float* w2  = (const float*)d_in[5];  const float* b2  = (const float*)d_in[6];
    const float* g2  = (const float*)d_in[7];  const float* be2 = (const float*)d_in[8];
    const float* w3  = (const float*)d_in[9];  const float* b3  = (const float*)d_in[10];
    const float* g3  = (const float*)d_in[11]; const float* be3 = (const float*)d_in[12];
    const float* w4  = (const float*)d_in[13]; const float* b4  = (const float*)d_in[14];
    const float* g4  = (const float*)d_in[15]; const float* be4 = (const float*)d_in[16];
    const float* w5  = (const float*)d_in[17]; const float* b5  = (const float*)d_in[18];
    const float* g5  = (const float*)d_in[19]; const float* be5 = (const float*)d_in[20];
    const float* w6  = (const float*)d_in[21]; const float* b6  = (const float*)d_in[22];
    const float* g6  = (const float*)d_in[23]; const float* be6 = (const float*)d_in[24];
    const float* w7  = (const float*)d_in[25]; const float* b7  = (const float*)d_in[26];
    const float* w8  = (const float*)d_in[27]; const float* b8  = (const float*)d_in[28];
    const float* w9  = (const float*)d_in[29]; const float* b9  = (const float*)d_in[30];
    float* out = (float*)d_out;

    char* ws = (char*)d_ws;
    size_t off = 0;
    auto alloc = [&](size_t bytes) { char* p = ws + off; off += (bytes + 255) & ~(size_t)255; return p; };

    const size_t P = PCOLS;
    float*    H     = (float*)   alloc(512 * P * 4);   // pre-BN f32 (reused; SP aliases it)
    _Float16* XA    = (_Float16*)alloc(320 * P * 2);   // ping: X0/X2/X4/X6
    _Float16* XB    = (_Float16*)alloc(512 * P * 2);   // pong: A8/X5/X7
    _Float16* X1    = (_Float16*)alloc( 64 * P * 2);   // pointfeat (persists)
    float*    G3    = (float*)   alloc(256 * P * 4);   // bn(conv3) f32, softpool source
    _Float16* Wh    = (_Float16*)alloc(1 << 20);       // packed small weights
    float*    Ypart = (float*)   alloc((size_t)16 * 256 * 512 * 4);  // conv8 split-K partials
    float*    Y8    = (float*)   alloc(256 * 512 * 4);
    int*      idx   = (int*)     alloc((size_t)BATCH * 256 * 32 * 4);
    float*    glob  = (float*)   alloc(BATCH * 256 * 4);
    float*    scale = (float*)   alloc(512 * 4);
    float*    shift = (float*)   alloc(512 * 4);
    _Float16* SP    = (_Float16*)ws;                   // alias H region (67 MB < 128 MB)
    _Float16* A8    = XB;                              // alias XB (33.5 MB < 64 MB)

    auto gemm = [&](const _Float16* A, const _Float16* Bm, float* C, const float* bias,
                    int M, int K) {
        dim3 grid((unsigned)(P / 128), (unsigned)(M / 64), 1);
        wmma_gemm_kernel<false><<<grid, 256, 0, stream>>>(A, Bm, C, bias, M, K, (int)P, K);
    };
    auto cvt = [&](const float* W, _Float16* D, int O, int C, int Kp) {
        cvt_pad_kernel<<<(unsigned)(((size_t)O * Kp + 255) / 256), 256, 0, stream>>>(W, D, O, C, Kp);
    };
    auto bnf16 = [&](int M, const float* gam, const float* bet, _Float16* dst) {
        bn_stats_kernel<<<M, 256, 0, stream>>>(H, gam, bet, scale, shift);
        bn_apply_kernel<true, true><<<(unsigned)(M * 256), 256, 0, stream>>>(H, scale, shift, dst);
    };

    // conv1 (K padded 4->32) -> BN+ReLU -> pointfeat X1
    pack_input_kernel<<<(unsigned)(32 * P / 256), 256, 0, stream>>>(x, XA);
    cvt(w1, Wh, 64, 4, 32);
    gemm(Wh, XA, H, b1, 64, 32);
    bnf16(64, g1, be1, X1);

    // conv2 -> BN+ReLU -> X2 (XA)
    cvt(w2, Wh, 128, 64, 64);
    gemm(Wh, X1, H, b2, 128, 64);
    bnf16(128, g2, be2, XA);

    // conv3 -> BN (no ReLU) -> G3 f32
    cvt(w3, Wh, 256, 128, 128);
    gemm(Wh, XA, H, b3, 256, 128);
    bn_stats_kernel<<<256, 256, 0, stream>>>(H, g3, be3, scale, shift);
    bn_apply_kernel<false, false><<<256 * 256, 256, 0, stream>>>(H, scale, shift, G3);

    // softpool: top-32 per (b, sort-channel), gather SP (65536 x 512) f16
    topk32_kernel<<<dim3(256, BATCH), 256, 0, stream>>>(G3, idx);
    gather_sp_kernel<<<(unsigned)(65536ULL * 512 / 256), 256, 0, stream>>>(G3, idx, SP);

    // conv8 as split-K WMMA GEMM: (256 x 65536) x (65536 x 512), 16 K-slices
    cvt(w8, A8, 256, 65536, 65536);
    {
        dim3 grid(512 / 128, 256 / 64, 16);
        wmma_gemm_kernel<true><<<grid, 256, 0, stream>>>(A8, SP, Ypart, nullptr,
                                                         256, 65536, 512, 4096);
        reduce_y8_kernel<<<512, 256, 0, stream>>>(Ypart, b8, Y8, 16);
    }

    // conv9 -> global feature (B x 256)
    conv9_kernel<<<dim3(256, BATCH), 256, 0, stream>>>(Y8, w9, b9, glob);

    // concat [glob | pointfeat] -> X4 (320 x P) in XA
    build_x4_kernel<<<(unsigned)(320 * P / 256), 256, 0, stream>>>(glob, X1, XA);

    // conv4 -> BN+ReLU -> X5 (XB)
    cvt(w4, Wh, 512, 320, 320);
    gemm(Wh, XA, H, b4, 512, 320);
    bnf16(512, g4, be4, XB);

    // conv5 -> BN+ReLU -> X6 (XA)
    cvt(w5, Wh, 256, 512, 512);
    gemm(Wh, XB, H, b5, 256, 512);
    bnf16(256, g5, be5, XA);

    // conv6 -> BN+ReLU -> X7 (XB)
    cvt(w6, Wh, 128, 256, 256);
    gemm(Wh, XA, H, b6, 128, 256);
    bnf16(128, g6, be6, XB);

    // conv7 + tanh -> (B,3,N)
    conv7_kernel<<<(unsigned)(P / 256), 256, 0, stream>>>(XB, w7, b7, out);
}
